// WindowAttention_58737972740780
// MI455X (gfx1250) — compile-verified, matrix operand order checked by reading, not run on hardware
//
#include <hip/hip_runtime.h>
#include <hip/hip_bf16.h>

typedef __attribute__((ext_vector_type(16))) _Float16 v16h;
typedef __attribute__((ext_vector_type(8)))  _Float16 v8h;
typedef __attribute__((ext_vector_type(4)))  _Float16 v4h;
typedef __attribute__((ext_vector_type(8)))  float    v8f;
typedef __attribute__((ext_vector_type(4)))  float    v4f;

#define NTOK 64
#define DIMC 192
#define NHEADS 6
#define QKVOUT 576
#define LOG100 4.6051701859880914f

// ---------------- workspace layout (bytes) ----------------
#define OFF_QKVW16   0          // 576*192 f16  = 221184
#define OFF_PROJW16  221184     // 192*192 f16  =  73728
#define OFF_QKVBIAS  294912     // 576 f32      =   2304
#define OFF_TBL      297216     // 225*6 f32    =   5400 (pad)
#define OFF_BIASB    302624     // 6*64*64 f32  =  98304
// total ws needed: 400928 bytes

// ---------------- LDS layout (dynamic, 245760 B) ----------------
// stage1:  [0,24576)        x16 (64x192 f16)
//          [24576,172032)   qkv f32 (64x576)
//          [172032,245760)  q16/k16/vT16 (3 * 6*64*32 f16)
// stage2+: [0,98304)        scores f32 (6*64*64)   (aliases x16+qkvf)
//          [98304,147456)   P16 f16 (6*64*64)      (aliases qkvf)
//          [147456,172032)  aout16 f16 (64x192)    (aliases qkvf)
#define LDS_BYTES 245760

__device__ __forceinline__ v16h load_frag(const _Float16* base, int stride, int lane) {
  // A/B fragment for v_wmma_f32_16x16x32_f16 (ISA 7.12.2 16-bit A layout):
  // lanes 0-15: K = [0..7],[16..23] of row=lane; lanes 16-31: K = [8..15],[24..31]
  const int row = lane & 15;
  const int c0  = (lane >> 4) << 3;
  const _Float16* p = base + row * stride + c0;
  union { v16h v; struct { v8h lo; v8h hi; } s; } u;
  u.s.lo = *(const v8h*)(p);
  u.s.hi = *(const v8h*)(p + 16);
  return u.v;
}

__device__ __forceinline__ v8f wmma16(v16h a, v16h b, v8f c) {
  return __builtin_amdgcn_wmma_f32_16x16x32_f16(false, a, false, b, (short)0, c, false, false);
}

// ================= prep kernels =================
__global__ void prep_weights(const float* __restrict__ qkv_w,
                             const float* __restrict__ proj_w,
                             const float* __restrict__ q_bias,
                             const float* __restrict__ v_bias,
                             _Float16* __restrict__ qkvw16,
                             _Float16* __restrict__ projw16,
                             float* __restrict__ qkvbias) {
  const int total = QKVOUT*DIMC + DIMC*DIMC + QKVOUT;
  for (int i = blockIdx.x * blockDim.x + threadIdx.x; i < total; i += gridDim.x * blockDim.x) {
    if (i < QKVOUT*DIMC) {
      qkvw16[i] = (_Float16)qkv_w[i];
    } else if (i < QKVOUT*DIMC + DIMC*DIMC) {
      int j = i - QKVOUT*DIMC;
      projw16[j] = (_Float16)proj_w[j];
    } else {
      int j = i - QKVOUT*DIMC - DIMC*DIMC;
      float v = (j < DIMC) ? q_bias[j] : ((j < 2*DIMC) ? 0.0f : v_bias[j - 2*DIMC]);
      qkvbias[j] = v;
    }
  }
}

__device__ __forceinline__ float cpb_coord(int i) {
  // ((i-7)/7 * 8) -> sign(v)*log2(|v|+1)/log2(8)
  float v = ((float)(i - 7)) * (8.0f / 7.0f);
  return copysignf(__log2f(fabsf(v) + 1.0f) * (1.0f / 3.0f), v);
}

__global__ void prep_cpb(const float* __restrict__ w1, const float* __restrict__ b1,
                         const float* __restrict__ w2, float* __restrict__ tbl) {
  __shared__ float hidden[512];
  const int t = blockIdx.x;           // 0..224
  const float c0 = cpb_coord(t / 15);
  const float c1 = cpb_coord(t % 15);
  for (int j = threadIdx.x; j < 512; j += blockDim.x)
    hidden[j] = fmaxf(c0 * w1[2*j] + c1 * w1[2*j + 1] + b1[j], 0.0f);
  __syncthreads();
  if (threadIdx.x < NHEADS) {
    const float* wr = w2 + threadIdx.x * 512;
    float s = 0.0f;
    for (int j = 0; j < 512; ++j) s += hidden[j] * wr[j];
    tbl[t * NHEADS + threadIdx.x] = s;
  }
}

__global__ void prep_bias(const float* __restrict__ tbl, float* __restrict__ biasb) {
  int e = blockIdx.x * blockDim.x + threadIdx.x;   // 6*64*64 = 24576 exact
  if (e >= NHEADS * NTOK * NTOK) return;
  int h = e >> 12;
  int p = e & 4095;
  int n = p >> 6, m = p & 63;
  int idx = ((n >> 3) - (m >> 3) + 7) * 15 + ((n & 7) - (m & 7) + 7);
  float r = tbl[idx * NHEADS + h];
  biasb[e] = 16.0f / (1.0f + __expf(-r));
}

// ================= fused attention kernel =================
__global__ __launch_bounds__(256) void swinv2_attn(
    const float* __restrict__ x, const float* __restrict__ mask,
    const float* __restrict__ logit_scale, const float* __restrict__ proj_b,
    const _Float16* __restrict__ qkvw16, const _Float16* __restrict__ projw16,
    const float* __restrict__ qkvbias, const float* __restrict__ biasb,
    float* __restrict__ out) {
  extern __shared__ char smem[];
  _Float16* x16    = (_Float16*)smem;
  float*    qkvf   = (float*)(smem + 24576);
  _Float16* q16    = (_Float16*)(smem + 172032);
  _Float16* k16    = q16 + NHEADS * 64 * 32;
  _Float16* vT     = k16 + NHEADS * 64 * 32;
  float*    scores = (float*)smem;
  _Float16* P16    = (_Float16*)(smem + 98304);
  _Float16* aout   = (_Float16*)(smem + 147456);

  const int tid  = threadIdx.x;
  const int wave = tid >> 5;
  const int lane = tid & 31;
  const int b    = blockIdx.x;
  const int colL = lane & 15;
  const int mh   = (lane >> 4) << 3;

  // ---- stage 0: x -> f16 LDS (vectorized: b128 loads, b64 LDS stores) ----
  {
    const v4f* xg4 = (const v4f*)(x + (size_t)b * (NTOK * DIMC));
    v4h* x164 = (v4h*)x16;
    for (int e = tid; e < (NTOK * DIMC) / 4; e += 256) {
      v4f t = xg4[e];
      v4h o;
#pragma unroll
      for (int u = 0; u < 4; ++u) o[u] = (_Float16)t[u];
      x164[e] = o;
    }
  }
  __syncthreads();

  // ---- stage 1: QKV = x @ W^T + bias (4 x 36 tiles, K=192, 2 tiles in flight) ----
  {
    const int r = wave & 3;
    const int cb = wave >> 2;
    v16h afr[6];
#pragma unroll
    for (int s = 0; s < 6; ++s)
      afr[s] = load_frag(x16 + r * 16 * DIMC + s * 32, DIMC, lane);
    for (int c = cb; c < 36; c += 4) {
      const int c2 = c + 2;
      v8f acc0 = {}, acc1 = {};
#pragma unroll
      for (int s = 0; s < 6; ++s) {
        v16h b0 = load_frag(qkvw16 + c  * 16 * DIMC + s * 32, DIMC, lane);
        v16h b1 = load_frag(qkvw16 + c2 * 16 * DIMC + s * 32, DIMC, lane);
        acc0 = wmma16(afr[s], b0, acc0);
        acc1 = wmma16(afr[s], b1, acc1);
      }
      const int col0 = c * 16 + colL, col1 = c2 * 16 + colL;
      const float bias0 = qkvbias[col0], bias1 = qkvbias[col1];
#pragma unroll
      for (int i = 0; i < 8; ++i) {
        qkvf[(r * 16 + mh + i) * QKVOUT + col0] = acc0[i] + bias0;
        qkvf[(r * 16 + mh + i) * QKVOUT + col1] = acc1[i] + bias1;
      }
    }
  }
  __syncthreads();

  // ---- stage 1b: cosine-normalize q,k -> f16; transpose v -> vT f16 ----
  for (int t = tid; t < 768; t += 256) {
    const int isk = t / 384;
    const int rem = t % 384;
    const int h = rem >> 6, row = rem & 63;
    const float* src = qkvf + row * QKVOUT + isk * DIMC + h * 32;
    float ss = 0.0f;
#pragma unroll
    for (int c = 0; c < 32; ++c) { float v = src[c]; ss += v * v; }
    const float inv = 1.0f / fmaxf(sqrtf(ss), 1e-12f);
    _Float16* dst = (isk ? k16 : q16) + h * 2048 + row * 32;
#pragma unroll
    for (int c = 0; c < 32; ++c) dst[c] = (_Float16)(src[c] * inv);
  }
  for (int e = tid; e < NTOK * DIMC; e += 256) {
    const int m = e / DIMC, col = e % DIMC;
    vT[(col >> 5) * 2048 + (col & 31) * 64 + m] =
        (_Float16)qkvf[m * QKVOUT + 2 * DIMC + col];
  }
  __syncthreads();

  // ---- stage 2a: S = qn@kn^T * scale + cpb + mask  (96 tiles over 8 waves) ----
  {
    const int win = b & 63;
#pragma unroll
    for (int j = 0; j < 12; ++j) {
      const int tt = wave + 8 * j;
      const int h = tt >> 4, tile = tt & 15;
      const int r = tile & 3, c = tile >> 2;
      v16h af = load_frag(q16 + h * 2048 + r * 16 * 32, 32, lane);
      v16h bf = load_frag(k16 + h * 2048 + c * 16 * 32, 32, lane);
      v8f acc = {};
      acc = wmma16(af, bf, acc);
      const float scale = __expf(fminf(logit_scale[h], LOG100));
#pragma unroll
      for (int i = 0; i < 8; ++i) {
        const int n = r * 16 + mh + i;
        const int m = c * 16 + colL;
        scores[h * 4096 + n * 64 + m] =
            acc[i] * scale + biasb[h * 4096 + n * 64 + m] + mask[win * 4096 + n * 64 + m];
      }
    }
  }
  __syncthreads();

  // ---- stage 2b: softmax rows -> P16 ----
  for (int t = tid; t < NHEADS * NTOK; t += 256) {
    const int h = t >> 6, n = t & 63;
    const float* srow = scores + h * 4096 + n * 64;
    float mx = -3.4e38f;
    for (int m = 0; m < 64; ++m) mx = fmaxf(mx, srow[m]);
    float sum = 0.0f;
    for (int m = 0; m < 64; ++m) sum += __expf(srow[m] - mx);
    const float rinv = 1.0f / sum;
    _Float16* prow = P16 + h * 4096 + n * 64;
    for (int m = 0; m < 64; ++m) prow[m] = (_Float16)(__expf(srow[m] - mx) * rinv);
  }
  __syncthreads();

  // ---- stage 2c: O_h = P @ v  (48 tiles over 8 waves, K=64) ----
  {
#pragma unroll
    for (int j = 0; j < 6; ++j) {
      const int tt = wave + 8 * j;
      const int h = tt >> 3, tile = tt & 7;
      const int r = tile & 3, c = tile >> 2;
      v8f acc = {};
#pragma unroll
      for (int s = 0; s < 2; ++s) {
        v16h af = load_frag(P16 + h * 4096 + r * 16 * 64 + s * 32, 64, lane);
        v16h bf = load_frag(vT + h * 2048 + c * 16 * 64 + s * 32, 64, lane);
        acc = wmma16(af, bf, acc);
      }
#pragma unroll
      for (int i = 0; i < 8; ++i)
        aout[(r * 16 + mh + i) * DIMC + h * 32 + c * 16 + colL] = (_Float16)acc[i];
    }
  }
  __syncthreads();

  // ---- stage 3: out = aout @ proj_w^T + proj_b (2 tiles in flight) ----
  {
    const int r = wave & 3;
    const int cb = wave >> 2;
    v16h afr[6];
#pragma unroll
    for (int s = 0; s < 6; ++s)
      afr[s] = load_frag(aout + r * 16 * DIMC + s * 32, DIMC, lane);
    float* og = out + (size_t)b * (NTOK * DIMC);
    for (int c = cb; c < 12; c += 4) {
      const int c2 = c + 2;
      v8f acc0 = {}, acc1 = {};
#pragma unroll
      for (int s = 0; s < 6; ++s) {
        v16h b0 = load_frag(projw16 + c  * 16 * DIMC + s * 32, DIMC, lane);
        v16h b1 = load_frag(projw16 + c2 * 16 * DIMC + s * 32, DIMC, lane);
        acc0 = wmma16(afr[s], b0, acc0);
        acc1 = wmma16(afr[s], b1, acc1);
      }
      const int col0 = c * 16 + colL, col1 = c2 * 16 + colL;
      const float pb0 = proj_b[col0], pb1 = proj_b[col1];
#pragma unroll
      for (int i = 0; i < 8; ++i) {
        og[(r * 16 + mh + i) * DIMC + col0] = acc0[i] + pb0;
        og[(r * 16 + mh + i) * DIMC + col1] = acc1[i] + pb1;
      }
    }
  }
}

extern "C" void kernel_launch(void* const* d_in, const int* in_sizes, int n_in,
                              void* d_out, int out_size, void* d_ws, size_t ws_size,
                              hipStream_t stream) {
  const float* x           = (const float*)d_in[0];
  const float* mask        = (const float*)d_in[1];
  const float* qkv_w       = (const float*)d_in[2];
  const float* q_bias      = (const float*)d_in[3];
  const float* v_bias      = (const float*)d_in[4];
  const float* logit_scale = (const float*)d_in[5];
  const float* cpb_w1      = (const float*)d_in[6];
  const float* cpb_b1      = (const float*)d_in[7];
  const float* cpb_w2      = (const float*)d_in[8];
  const float* proj_w      = (const float*)d_in[9];
  const float* proj_b      = (const float*)d_in[10];

  char* ws = (char*)d_ws;
  _Float16* qkvw16  = (_Float16*)(ws + OFF_QKVW16);
  _Float16* projw16 = (_Float16*)(ws + OFF_PROJW16);
  float*    qkvbias = (float*)(ws + OFF_QKVBIAS);
  float*    tbl     = (float*)(ws + OFF_TBL);
  float*    biasb   = (float*)(ws + OFF_BIASB);

  const int nB = in_sizes[0] / (NTOK * DIMC);  // 4096 windows

  prep_weights<<<580, 256, 0, stream>>>(qkv_w, proj_w, q_bias, v_bias,
                                        qkvw16, projw16, qkvbias);
  prep_cpb<<<225, 256, 0, stream>>>(cpb_w1, cpb_b1, cpb_w2, tbl);
  prep_bias<<<(NHEADS * NTOK * NTOK + 255) / 256, 256, 0, stream>>>(tbl, biasb);
  swinv2_attn<<<nB, 256, LDS_BYTES, stream>>>(x, mask, logit_scale, proj_b,
                                              qkvw16, projw16, qkvbias, biasb,
                                              (float*)d_out);
}
